// SchNetChemicalShiftHead_20426864459959
// MI455X (gfx1250) — matrix operand order repaired and checked by reading, not compiled, not run on hardware
//
#include <hip/hip_runtime.h>
#include <hip/hip_bf16.h>

typedef __attribute__((ext_vector_type(16))) _Float16 v16h;
typedef __attribute__((ext_vector_type(8)))  _Float16 v8h;
typedef __attribute__((ext_vector_type(8)))  float    v8f;

#define NG 50
#define CUTOFF 10.0f
#define LOG2F_ 0.69314718055994531f

__device__ __forceinline__ float sspf(float x) {
  // shifted softplus, numerically stable
  return fmaxf(x, 0.0f) + log1pf(__expf(-fabsf(x))) - LOG2F_;
}

// Load 16x32 f16 A-fragment from row-major storage (stride in halves).
// Lane L: M = base row + (L&15), halves 0..7 <-> K=kb..kb+7, halves 8..15 <-> K=kb+16..kb+23,
// kb = kstep*32 + (L>>4)*8.  Two 16-byte loads per lane.
__device__ __forceinline__ v16h load_a_frag(const _Float16* base, int stride, int m,
                                            int khalves, int lane) {
  int kb = khalves + ((lane >> 4) * 8);
  const _Float16* p = base + m * stride + kb;
  v8h lo = *(const v8h*)p;
  v8h hi = *(const v8h*)(p + 16);
  v16h r;
#pragma unroll
  for (int e = 0; e < 8; ++e) { r[e] = lo[e]; r[e + 8] = hi[e]; }
  return r;
}

// ---- weight pre-pack: row-major f32 [K x 128] -> f16 B-fragment layout ----
// fragment half index: (((nt*ksteps + ks)*32) + lane)*16 + h
// element (lane,h) = W[ks*32 + (lane>>4)*8 + (h&7) + (h>>3)*16][nt*16 + (lane&15)]
__global__ void pack_b_kernel(const float* __restrict__ src, _Float16* __restrict__ dst,
                              int kreal, int ksteps) {
  int idx = blockIdx.x * blockDim.x + threadIdx.x;
  int total = 8 * ksteps * 512;
  if (idx >= total) return;
  int per_nt = ksteps * 512;
  int nt = idx / per_nt;
  int r  = idx - nt * per_nt;
  int ks = r >> 9;
  int r2 = r & 511;
  int lane = r2 >> 4;
  int hh = r2 & 15;
  int k = ks * 32 + ((lane >> 4) * 8) + (hh & 7) + ((hh >> 3) * 16);
  int n = nt * 16 + (lane & 15);
  float v = (k < kreal) ? src[k * 128 + n] : 0.0f;
  dst[idx] = (_Float16)v;
}

// ---- embedding gather ----
__global__ void embed_kernel(const int* __restrict__ z, const float* __restrict__ emb,
                             float* __restrict__ h, _Float16* __restrict__ h16, int n) {
  int idx = blockIdx.x * blockDim.x + threadIdx.x;
  if (idx >= n * 128) return;
  int i = idx >> 7, f = idx & 127;
  float v = emb[z[i] * 128 + f];
  h[idx] = v;
  h16[idx] = (_Float16)v;
}

// ---- generic [N x 128] x [128 x 128] WMMA GEMM, 64 rows / block, 4 waves ----
// MODE 0: out16 = f16(A@B)           (lin1, no bias)
// MODE 1: out16 = f16(ssp(A@B + b))  (lin2 + shifted softplus)
// MODE 2: h += A@B + b; h16 = f16(h) (int_lin + residual)
template <int MODE>
__global__ __launch_bounds__(128) void gemm_kernel(
    const _Float16* __restrict__ A, const _Float16* __restrict__ Bp,
    const float* __restrict__ bias, _Float16* __restrict__ out16,
    float* __restrict__ hio, _Float16* __restrict__ h16o, int nrows) {
  const int tid = threadIdx.x;
  const int lane = tid & 31;
  const int wave = tid >> 5;
  const int rowBase = blockIdx.x * 64;
  v8f acc[4][2] = {};
#pragma unroll
  for (int ks = 0; ks < 4; ++ks) {
    v16h a[4];
#pragma unroll
    for (int mt = 0; mt < 4; ++mt) {
      int m = rowBase + mt * 16 + (lane & 15);
      if (m >= nrows) m = nrows - 1;
      a[mt] = load_a_frag(A, 128, m, ks * 32, lane);
    }
#pragma unroll
    for (int ntl = 0; ntl < 2; ++ntl) {
      int nt = wave * 2 + ntl;
      v16h b = *((const v16h*)Bp + (nt * 4 + ks) * 32 + lane);
#pragma unroll
      for (int mt = 0; mt < 4; ++mt)
        acc[mt][ntl] = __builtin_amdgcn_wmma_f32_16x16x32_f16(
            false, a[mt], false, b, (short)0, acc[mt][ntl], false, false);
    }
  }
#pragma unroll
  for (int ntl = 0; ntl < 2; ++ntl) {
    int n = (wave * 2 + ntl) * 16 + (lane & 15);
    float bv = (MODE == 0) ? 0.0f : bias[n];
#pragma unroll
    for (int mt = 0; mt < 4; ++mt) {
#pragma unroll
      for (int v = 0; v < 8; ++v) {
        int m = rowBase + mt * 16 + ((lane >> 4) * 8) + v;
        if (m < nrows) {
          float val = acc[mt][ntl][v] + bv;
          if (MODE == 1) val = sspf(val);
          if (MODE == 2) {
            float hn = hio[m * 128 + n] + val;
            hio[m * 128 + n] = hn;
            h16o[m * 128 + n] = (_Float16)hn;
          } else {
            out16[m * 128 + n] = (_Float16)val;
          }
        }
      }
    }
  }
}

// ---- fused edge kernel: RBF -> filter MLP (WMMA) -> cutoff -> gather*W -> segment-sum ----
// 2 atoms (64 contiguous edges, row = repeat(arange(N),32)) per 128-thread workgroup.
// M=64 amortizes B-fragment (L2) traffic over 4 M-tiles and halves barrier overhead/atom.
__global__ __launch_bounds__(128) void edge_kernel(
    const float* __restrict__ pos, const int* __restrict__ col,
    const _Float16* __restrict__ X16,
    const _Float16* __restrict__ w1p, const _Float16* __restrict__ w2p,
    const float* __restrict__ b1, const float* __restrict__ b2,
    _Float16* __restrict__ agg16, int natoms) {
  const int atomBase = blockIdx.x * 2;
  const int tid = threadIdx.x;
  const int lane = tid & 31;
  const int wave = tid >> 5;

  __shared__ int      sCol[64];
  __shared__ float    sDist[64];
  __shared__ float    sC[64];
  __shared__ _Float16 sA[64 * 64];    // padded RBF rows (A of GEMM1)      8 KB
  __shared__ _Float16 sT[64 * 128];   // ssp(rbf@W1+b1)   (A of GEMM2)    16 KB
  __shared__ float    sW[64 * 128];   // final edge filter                32 KB

  if (tid < 64) {
    int a = atomBase + (tid >> 5);
    if (a >= natoms) a = natoms - 1;
    int c = col[a * 32 + (tid & 31)];
    sCol[tid] = c;
    float dx = pos[a * 3 + 0] - pos[c * 3 + 0];
    float dy = pos[a * 3 + 1] - pos[c * 3 + 1];
    float dz = pos[a * 3 + 2] - pos[c * 3 + 2];
    float d = sqrtf(dx * dx + dy * dy + dz * dz);
    sDist[tid] = d;
    sC[tid] = 0.5f * (__cosf(d * (3.14159265358979f / CUTOFF)) + 1.0f);
  }
  __syncthreads();
  {
    const float step = CUTOFF / (float)(NG - 1);
    const float coeff = -0.5f / (step * step);
    int e = tid & 63;
    int gb = (tid >> 6) * 32;
    float d = sDist[e];
#pragma unroll
    for (int j = 0; j < 32; ++j) {
      int g = gb + j;
      float t = d - step * (float)g;
      float v = (g < NG) ? __expf(coeff * t * t) : 0.0f;
      sA[e * 64 + g] = (_Float16)v;
    }
  }
  __syncthreads();

  // GEMM1: [64x64] @ [64x128] + b1 -> ssp -> sT
  {
    v8f acc[4][2] = {};
#pragma unroll
    for (int ks = 0; ks < 2; ++ks) {
      v16h a[4];
#pragma unroll
      for (int mt = 0; mt < 4; ++mt)
        a[mt] = load_a_frag(sA, 64, mt * 16 + (lane & 15), ks * 32, lane);
#pragma unroll
      for (int ntl = 0; ntl < 2; ++ntl) {
        int nt = wave * 2 + ntl;
        v16h b = *((const v16h*)w1p + (nt * 2 + ks) * 32 + lane);
#pragma unroll
        for (int mt = 0; mt < 4; ++mt)
          acc[mt][ntl] = __builtin_amdgcn_wmma_f32_16x16x32_f16(
              false, a[mt], false, b, (short)0, acc[mt][ntl], false, false);
      }
    }
#pragma unroll
    for (int ntl = 0; ntl < 2; ++ntl) {
      int n = (wave * 2 + ntl) * 16 + (lane & 15);
      float bv = b1[n];
#pragma unroll
      for (int mt = 0; mt < 4; ++mt)
#pragma unroll
        for (int v = 0; v < 8; ++v) {
          int m = mt * 16 + ((lane >> 4) * 8) + v;
          sT[m * 128 + n] = (_Float16)sspf(acc[mt][ntl][v] + bv);
        }
    }
  }
  __syncthreads();

  // GEMM2: ([64x128] @ [128x128] + b2) * C -> sW
  {
    v8f acc[4][2] = {};
#pragma unroll
    for (int ks = 0; ks < 4; ++ks) {
      v16h a[4];
#pragma unroll
      for (int mt = 0; mt < 4; ++mt)
        a[mt] = load_a_frag(sT, 128, mt * 16 + (lane & 15), ks * 32, lane);
#pragma unroll
      for (int ntl = 0; ntl < 2; ++ntl) {
        int nt = wave * 2 + ntl;
        v16h b = *((const v16h*)w2p + (nt * 4 + ks) * 32 + lane);
#pragma unroll
        for (int mt = 0; mt < 4; ++mt)
          acc[mt][ntl] = __builtin_amdgcn_wmma_f32_16x16x32_f16(
              false, a[mt], false, b, (short)0, acc[mt][ntl], false, false);
      }
    }
#pragma unroll
    for (int ntl = 0; ntl < 2; ++ntl) {
      int n = (wave * 2 + ntl) * 16 + (lane & 15);
      float bv = b2[n];
#pragma unroll
      for (int mt = 0; mt < 4; ++mt)
#pragma unroll
        for (int v = 0; v < 8; ++v) {
          int m = mt * 16 + ((lane >> 4) * 8) + v;
          sW[m * 128 + n] = (acc[mt][ntl][v] + bv) * sC[m];
        }
    }
  }
  __syncthreads();

  // agg[atom][f] = sum_e X16[col[e]][f] * W[e][f]  (coalesced gather, no atomics)
  {
    int f = tid;
#pragma unroll
    for (int a = 0; a < 2; ++a) {
      int atom = atomBase + a;
      if (atom >= natoms) break;
      float acc = 0.0f;
#pragma unroll 4
      for (int e = a * 32; e < a * 32 + 32; ++e) {
        int c = sCol[e];
        acc += (float)X16[c * 128 + f] * sW[e * 128 + f];
      }
      agg16[atom * 128 + f] = (_Float16)acc;
    }
  }
}

// ---- prediction head: 128 -> 64 -> 32 -> 1 with PReLU ----
__global__ __launch_bounds__(64) void head_kernel(
    const float* __restrict__ h, const float* __restrict__ w1, const float* __restrict__ b1,
    const float* __restrict__ a1, const float* __restrict__ w2, const float* __restrict__ b2,
    const float* __restrict__ a2, const float* __restrict__ w3, const float* __restrict__ b3,
    float* __restrict__ out) {
  int atom = blockIdx.x;
  int tid = threadIdx.x;
  __shared__ float s1[64];
  __shared__ float s2[32];
  {
    float acc = b1[tid];
    const float* hr = h + atom * 128;
    for (int k = 0; k < 128; ++k) acc += hr[k] * w1[k * 64 + tid];
    float al = a1[0];
    s1[tid] = (acc >= 0.f) ? acc : al * acc;
  }
  __syncthreads();
  if (tid < 32) {
    float acc = b2[tid];
    for (int k = 0; k < 64; ++k) acc += s1[k] * w2[k * 32 + tid];
    float al = a2[0];
    s2[tid] = ((acc >= 0.f) ? acc : al * acc) * w3[tid];
  }
  __syncthreads();
  if (tid == 0) {
    float acc = b3[0];
    for (int k = 0; k < 32; ++k) acc += s2[k];
    out[atom] = acc;
  }
}

extern "C" void kernel_launch(void* const* d_in, const int* in_sizes, int n_in,
                              void* d_out, int out_size, void* d_ws, size_t ws_size,
                              hipStream_t stream) {
  (void)n_in; (void)out_size; (void)ws_size;
  const int*   z          = (const int*)d_in[0];
  const float* pos        = (const float*)d_in[1];
  const int*   edge_index = (const int*)d_in[2];
  const float* emb        = (const float*)d_in[3];
  const float* mlp_w1     = (const float*)d_in[4];
  const float* mlp_b1     = (const float*)d_in[5];
  const float* mlp_w2     = (const float*)d_in[6];
  const float* mlp_b2     = (const float*)d_in[7];
  const float* lin1_w     = (const float*)d_in[8];
  const float* lin2_w     = (const float*)d_in[9];
  const float* lin2_b     = (const float*)d_in[10];
  const float* int_lin_w  = (const float*)d_in[11];
  const float* int_lin_b  = (const float*)d_in[12];
  const float* head_w1    = (const float*)d_in[13];
  const float* head_b1    = (const float*)d_in[14];
  const float* a1         = (const float*)d_in[15];
  const float* head_w2    = (const float*)d_in[16];
  const float* head_b2    = (const float*)d_in[17];
  const float* a2         = (const float*)d_in[18];
  const float* head_w3    = (const float*)d_in[19];
  const float* head_b3    = (const float*)d_in[20];

  const int N = in_sizes[0];
  const int E = in_sizes[2] / 2;
  const int* col = edge_index + E;
  const int L = 6;

  char* ws = (char*)d_ws;
  size_t off = 0;
  auto alloc = [&](size_t bytes) -> void* {
    void* p = ws + off;
    off = (off + bytes + 255) & ~size_t(255);
    return p;
  };
  float*    h     = (float*)alloc((size_t)N * 128 * 4);
  _Float16* h16   = (_Float16*)alloc((size_t)N * 128 * 2);
  _Float16* X16   = (_Float16*)alloc((size_t)N * 128 * 2);
  _Float16* agg16 = (_Float16*)alloc((size_t)N * 128 * 2);
  _Float16* T16   = X16;  // X16 dead after edge_kernel; reuse for ssp(lin2) output
  _Float16 *w1p[6], *w2p[6], *l1p[6], *l2p[6], *ilp[6];
  for (int l = 0; l < L; ++l) {
    w1p[l] = (_Float16*)alloc(8 * 2 * 512 * 2);
    w2p[l] = (_Float16*)alloc(8 * 4 * 512 * 2);
    l1p[l] = (_Float16*)alloc(8 * 4 * 512 * 2);
    l2p[l] = (_Float16*)alloc(8 * 4 * 512 * 2);
    ilp[l] = (_Float16*)alloc(8 * 4 * 512 * 2);
  }

  for (int l = 0; l < L; ++l) {
    pack_b_kernel<<<(8 * 2 * 512 + 255) / 256, 256, 0, stream>>>(mlp_w1 + (size_t)l * 50 * 128, w1p[l], 50, 2);
    pack_b_kernel<<<(8 * 4 * 512 + 255) / 256, 256, 0, stream>>>(mlp_w2 + (size_t)l * 128 * 128, w2p[l], 128, 4);
    pack_b_kernel<<<(8 * 4 * 512 + 255) / 256, 256, 0, stream>>>(lin1_w + (size_t)l * 128 * 128, l1p[l], 128, 4);
    pack_b_kernel<<<(8 * 4 * 512 + 255) / 256, 256, 0, stream>>>(lin2_w + (size_t)l * 128 * 128, l2p[l], 128, 4);
    pack_b_kernel<<<(8 * 4 * 512 + 255) / 256, 256, 0, stream>>>(int_lin_w + (size_t)l * 128 * 128, ilp[l], 128, 4);
  }

  embed_kernel<<<(N * 128 + 255) / 256, 256, 0, stream>>>(z, emb, h, h16, N);

  const int gblocks = (N + 63) / 64;
  const int eblocks = (N + 1) / 2;
  for (int l = 0; l < L; ++l) {
    gemm_kernel<0><<<gblocks, 128, 0, stream>>>(h16, l1p[l], nullptr, X16, nullptr, nullptr, N);
    edge_kernel<<<eblocks, 128, 0, stream>>>(pos, col, X16, w1p[l], w2p[l],
                                             mlp_b1 + l * 128, mlp_b2 + l * 128, agg16, N);
    gemm_kernel<1><<<gblocks, 128, 0, stream>>>(agg16, l2p[l], lin2_b + l * 128, T16, nullptr, nullptr, N);
    gemm_kernel<2><<<gblocks, 128, 0, stream>>>(T16, ilp[l], int_lin_b + l * 128, nullptr, h, h16, N);
  }

  head_kernel<<<N, 64, 0, stream>>>(h, head_w1, head_b1, a1, head_w2, head_b2, a2,
                                    head_w3, head_b3, (float*)d_out);
}
